// Attention_ori_45191645888733
// MI455X (gfx1250) — matrix operand order repaired and checked by reading, not compiled, not run on hardware
//
#include <hip/hip_runtime.h>
#include <hip/hip_bf16.h>

// ---------------------------------------------------------------------------
// Problem constants (match reference)
// ---------------------------------------------------------------------------
#define HH   8
#define DD   64
#define DIM  512
#define NN   1024
#define BB   8
#define ROWS (BB * NN)        // 8192
#define QKVC (3 * DIM)        // 1536
#define LN_EPS 1e-5f

typedef __attribute__((ext_vector_type(16))) __bf16 bf16x16;
typedef __attribute__((ext_vector_type(8)))  __bf16 bf16x8;
typedef __attribute__((ext_vector_type(8)))  float  f32x8;

// ---------------------------------------------------------------------------
// WMMA fragment helpers (wave32; A/B use this code's contiguous-K order, C/D per CDNA5 ISA §7.12.2)
// A 16x32 bf16: lane L -> row L&15; elems = 16 contiguous K from koff,
//               koff = (L>=16)*16 -> two 16B loads from row-major storage.
// B 32x16 bf16: lane L -> col L&15; elems = 16 contiguous K from (L>=16)*16
//               -> one 32B load from K-contiguous-per-column storage.
// C/D f32 16x16: VGPR r -> row r + (L>=16)*8, col = L&15.
// ---------------------------------------------------------------------------
__device__ __forceinline__ bf16x16 load_a_frag(const __bf16* base, int ld,
                                               int row0, int kk) {
  int lane = threadIdx.x & 31;
  int m    = lane & 15;
  int k8   = (lane >> 4) << 4;            // 0 or 16
  const __bf16* p = base + (size_t)(row0 + m) * ld + kk + k8;
  union { bf16x16 v; bf16x8 h[2]; } u;
  u.h[0] = *(const bf16x8*)(p);
  u.h[1] = *(const bf16x8*)(p + 8);
  return u.v;
}

__device__ __forceinline__ bf16x16 load_b_frag(const __bf16* base, int ld,
                                               int col0, int kk) {
  int lane = threadIdx.x & 31;
  int n    = lane & 15;
  int k16  = (lane >> 4) << 4;            // 0 or 16
  return *(const bf16x16*)(base + (size_t)(col0 + n) * ld + kk + k16);
}

__device__ __forceinline__ f32x8 wmma_bf16(bf16x16 a, bf16x16 b, f32x8 c) {
  return __builtin_amdgcn_wmma_f32_16x16x32_bf16(false, a, false, b,
                                                 (short)0, c, false, false);
}

__device__ __forceinline__ float rmax16(float v) {
  v = fmaxf(v, __shfl_xor(v, 1, 32));
  v = fmaxf(v, __shfl_xor(v, 2, 32));
  v = fmaxf(v, __shfl_xor(v, 4, 32));
  v = fmaxf(v, __shfl_xor(v, 8, 32));
  return v;
}
__device__ __forceinline__ float rsum16(float v) {
  v += __shfl_xor(v, 1, 32);
  v += __shfl_xor(v, 2, 32);
  v += __shfl_xor(v, 4, 32);
  v += __shfl_xor(v, 8, 32);
  return v;
}

// ---------------------------------------------------------------------------
// Prep kernels: fp32 -> bf16 (+ weight transposes so B-fragments are 32B loads)
// ---------------------------------------------------------------------------
__global__ void k_prep_x(const float* __restrict__ x, __bf16* __restrict__ xbf) {
  int i = blockIdx.x * 256 + threadIdx.x;          // ROWS*DIM
  xbf[i] = (__bf16)x[i];
}

__global__ void k_prep_wqkv(const float* __restrict__ w, __bf16* __restrict__ wT) {
  int i = blockIdx.x * 256 + threadIdx.x;          // DIM*QKVC
  int k = i / QKVC, n = i % QKVC;                  // coalesced read
  wT[(size_t)n * DIM + k] = (__bf16)w[i];          // wT[out][in]
}

__global__ void k_prep_wout(const float* __restrict__ w, __bf16* __restrict__ wT) {
  int i = blockIdx.x * 256 + threadIdx.x;          // DIM*DIM
  int k = i >> 9, n = i & 511;
  wT[(size_t)n * DIM + k] = (__bf16)w[i];
}

// ---------------------------------------------------------------------------
// Copy-free pipelined 16x64 GEMM tile body: fully-unrolled ping-pong over K.
// Every buffer assignment is an SSA rename (no v_mov rotation); loads for the
// next 32-wide K step are in flight while the current 4 WMMAs issue.
// ---------------------------------------------------------------------------
__device__ __forceinline__ void gemm_tile_16x64(const __bf16* __restrict__ abase,
                                                const __bf16* __restrict__ wT,
                                                int row0, int col0, f32x8 acc[4]) {
  bf16x16 aA = load_a_frag(abase, DIM, row0, 0);
  bf16x16 bA[4];
#pragma unroll
  for (int t = 0; t < 4; ++t) bA[t] = load_b_frag(wT, DIM, col0 + t * 16, 0);

#pragma unroll
  for (int kk = 0; kk < DIM; kk += 64) {
    bf16x16 aB = load_a_frag(abase, DIM, row0, kk + 32);
    bf16x16 bB[4];
#pragma unroll
    for (int t = 0; t < 4; ++t) bB[t] = load_b_frag(wT, DIM, col0 + t * 16, kk + 32);
#pragma unroll
    for (int t = 0; t < 4; ++t) acc[t] = wmma_bf16(aA, bA[t], acc[t]);

    if (kk + 64 < DIM) {                            // compile-time under unroll
      aA = load_a_frag(abase, DIM, row0, kk + 64);
#pragma unroll
      for (int t = 0; t < 4; ++t) bA[t] = load_b_frag(wT, DIM, col0 + t * 16, kk + 64);
    }
#pragma unroll
    for (int t = 0; t < 4; ++t) acc[t] = wmma_bf16(aB, bB[t], acc[t]);
  }
}

// ---------------------------------------------------------------------------
// GEMM1: qkv[8192,1536] = x_bf @ w_qkv.  Wave computes 16x64.
// ---------------------------------------------------------------------------
__global__ __launch_bounds__(256) void k_gemm_qkv(const __bf16* __restrict__ xbf,
                                                  const __bf16* __restrict__ wT,
                                                  __bf16* __restrict__ qkv) {
  int wid  = blockIdx.x * 8 + (threadIdx.x >> 5);  // 12288 wave tiles
  int rt   = wid / 24;                             // 512 row tiles
  int cg   = wid % 24;                             // 24 col groups of 64
  int lane = threadIdx.x & 31;
  int row0 = rt * 16, col0 = cg * 64;

  f32x8 acc[4] = {};
  gemm_tile_16x64(xbf, wT, row0, col0, acc);

  int n = lane & 15, rofs = (lane >> 4) << 3;
#pragma unroll
  for (int t = 0; t < 4; ++t)
#pragma unroll
    for (int r = 0; r < 8; ++r)
      qkv[(size_t)(row0 + r + rofs) * QKVC + col0 + t * 16 + n] = (__bf16)acc[t][r];
}

// ---------------------------------------------------------------------------
// Repack K (with the reference's row-major reshape quirk) and V for B-fragments.
//   krT[g][j][a] = k_r[g][a][j] = qkv[b*N + 16a + j/64][512 + h*64 + (j&63)]
//   vT [g][d][j] = v[g][j][d]   = qkv[b*N + j][1024 + h*64 + d]
// ---------------------------------------------------------------------------
__global__ void k_repack_kv(const __bf16* __restrict__ qkv,
                            __bf16* __restrict__ krT, __bf16* __restrict__ vT) {
  int i   = blockIdx.x * 256 + threadIdx.x;        // 64*1024*64
  int g   = i >> 16;
  int rem = i & 65535;
  int b   = g >> 3, h = g & 7;

  int j = rem >> 6, a = rem & 63;                  // krT: [j][a]
  krT[i] = qkv[(size_t)(b * NN + a * 16 + (j >> 6)) * QKVC + DIM + h * DD + (j & 63)];

  int d = rem >> 10, jj = rem & 1023;              // vT: [d][j]
  vT[i] = qkv[(size_t)(b * NN + jj) * QKVC + 2 * DIM + h * DD + d];
}

// ---------------------------------------------------------------------------
// Flash attention: wave owns head g, 16 query rows; streams 1024 keys in 32-wide
// steps with online softmax.  All 8 B-fragments (Kr + V) are issued up front so
// the V loads are hidden under the softmax VALU work and the LDS transpose.
// ---------------------------------------------------------------------------
__global__ __launch_bounds__(128) void k_attn(const __bf16* __restrict__ qkv,
                                              const __bf16* __restrict__ krT,
                                              const __bf16* __restrict__ vT,
                                              __bf16* __restrict__ aoBf) {
  __shared__ __align__(32) __bf16 ldsP[4][16 * 32];
  int wib  = threadIdx.x >> 5;
  int wid  = blockIdx.x * 4 + wib;                 // 4096 waves
  int g    = wid >> 6;                             // head 0..63
  int qt   = wid & 63;                             // q tile 0..63
  int b    = g >> 3, h = g & 7;
  int lane = threadIdx.x & 31;
  __bf16* pbuf = ldsP[wib];

  const __bf16* qbase = qkv + (size_t)(b * NN) * QKVC + h * DD;   // ld=1536
  bf16x16 aq0 = load_a_frag(qbase, QKVC, qt * 16, 0);
  bf16x16 aq1 = load_a_frag(qbase, QKVC, qt * 16, 32);

  const __bf16* krg = krT + (size_t)g * NN * DD;   // [j][a], ld=64
  const __bf16* vg  = vT  + (size_t)g * DD * NN;   // [d][j], ld=1024

  f32x8 o0 = {}, o1 = {}, o2 = {}, o3 = {};
  float m8[8], l8[8];
#pragma unroll
  for (int r = 0; r < 8; ++r) { m8[r] = -1e30f; l8[r] = 0.f; }

  int n = lane & 15, rofs = (lane >> 4) << 3;

  for (int j0 = 0; j0 < NN; j0 += 32) {
    if (j0 + 32 < NN) {
      __builtin_prefetch(krg + (size_t)(j0 + 32) * DD, 0, 0);
      __builtin_prefetch(vg + j0 + 32, 0, 0);
    }
    // Issue ALL loads for this step first: 4 Kr tiles + 4 V tiles (one clause).
    bf16x16 bk0 = load_b_frag(krg, DD, j0,      0);
    bf16x16 bk1 = load_b_frag(krg, DD, j0,      32);
    bf16x16 bk2 = load_b_frag(krg, DD, j0 + 16, 0);
    bf16x16 bk3 = load_b_frag(krg, DD, j0 + 16, 32);
    bf16x16 bv0 = load_b_frag(vg, NN, 0,  j0);
    bf16x16 bv1 = load_b_frag(vg, NN, 16, j0);
    bf16x16 bv2 = load_b_frag(vg, NN, 32, j0);
    bf16x16 bv3 = load_b_frag(vg, NN, 48, j0);

    // S tiles: 16 x 32 scores, K = 64 contraction (2 WMMAs each)
    f32x8 s0 = {}, s1 = {};
    s0 = wmma_bf16(aq0, bk0, s0);
    s0 = wmma_bf16(aq1, bk1, s0);
    s1 = wmma_bf16(aq0, bk2, s1);
    s1 = wmma_bf16(aq1, bk3, s1);

    // online softmax (row stats share the C layout: slot r -> row r+rofs)
#pragma unroll
    for (int r = 0; r < 8; ++r) {
      float tmax = rmax16(fmaxf(s0[r], s1[r]));
      float mn   = fmaxf(m8[r], tmax);
      float sc   = __expf(m8[r] - mn);
      float e0   = __expf(s0[r] - mn);
      float e1   = __expf(s1[r] - mn);
      l8[r] = l8[r] * sc + rsum16(e0 + e1);
      m8[r] = mn;
      s0[r] = e0; s1[r] = e1;
      o0[r] *= sc; o1[r] *= sc; o2[r] *= sc; o3[r] *= sc;
    }

    // C-layout P -> row-major LDS -> A-layout fragment
#pragma unroll
    for (int r = 0; r < 8; ++r) {
      pbuf[(r + rofs) * 32 + n]      = (__bf16)s0[r];
      pbuf[(r + rofs) * 32 + 16 + n] = (__bf16)s1[r];
    }
    asm volatile("s_wait_dscnt 0x0" ::: "memory");
    bf16x16 pa;
    {
      int koff = (lane >> 4) << 4;                 // 0 or 16
      union { bf16x16 v; bf16x8 h[2]; } u;
      u.h[0] = *(const bf16x8*)(pbuf + n * 32 + koff);
      u.h[1] = *(const bf16x8*)(pbuf + n * 32 + koff + 8);
      pa = u.v;
    }
    asm volatile("" ::: "memory");                 // keep reads before next stores

    // O += P @ V  (4 d-column tiles; V fragments already resident)
    o0 = wmma_bf16(pa, bv0, o0);
    o1 = wmma_bf16(pa, bv1, o1);
    o2 = wmma_bf16(pa, bv2, o2);
    o3 = wmma_bf16(pa, bv3, o3);
  }

  // normalize and store into '(b h) n d -> b n (h d)' bf16 layout
#pragma unroll
  for (int r = 0; r < 8; ++r) {
    float inv = 1.0f / l8[r];
    __bf16* p = aoBf + (size_t)(b * NN + qt * 16 + r + rofs) * DIM + h * DD;
    p[n]      = (__bf16)(o0[r] * inv);
    p[16 + n] = (__bf16)(o1[r] * inv);
    p[32 + n] = (__bf16)(o2[r] * inv);
    p[48 + n] = (__bf16)(o3[r] * inv);
  }
}

// ---------------------------------------------------------------------------
// GEMM3 fused with bias + LayerNorm + residual.
// Block (8 waves) computes 16 full rows (16x512) into LDS, then LN per row.
// ---------------------------------------------------------------------------
__global__ __launch_bounds__(256) void k_proj_ln(const __bf16* __restrict__ aoBf,
                                                 const __bf16* __restrict__ wT,
                                                 const float* __restrict__ bout,
                                                 const float* __restrict__ lng,
                                                 const float* __restrict__ lnb,
                                                 const float* __restrict__ x,
                                                 float* __restrict__ out) {
  __shared__ float ldsO[16 * DIM];                 // 32 KB
  int w    = threadIdx.x >> 5;
  int lane = threadIdx.x & 31;
  int row0 = blockIdx.x * 16;
  int col0 = w * 64;

  f32x8 acc[4] = {};
  gemm_tile_16x64(aoBf, wT, row0, col0, acc);

  int n = lane & 15, rofs = (lane >> 4) << 3;
#pragma unroll
  for (int t = 0; t < 4; ++t)
#pragma unroll
    for (int r = 0; r < 8; ++r) {
      int c = col0 + t * 16 + n;
      ldsO[(r + rofs) * DIM + c] = acc[t][r] + bout[c];
    }
  __syncthreads();

  // LayerNorm + residual: 16 lanes per row, 32 cols per lane
  int row = threadIdx.x >> 4;
  int l16 = threadIdx.x & 15;
  float s = 0.f, ss = 0.f;
#pragma unroll 8
  for (int j = 0; j < 32; ++j) {
    float v = ldsO[row * DIM + l16 + j * 16];
    s += v; ss += v * v;
  }
  s = rsum16(s); ss = rsum16(ss);
  float mu   = s * (1.0f / DIM);
  float var  = ss * (1.0f / DIM) - mu * mu;
  float rstd = rsqrtf(var + LN_EPS);
  const float* xr = x + (size_t)(row0 + row) * DIM;
  float* orow = out + (size_t)(row0 + row) * DIM;
#pragma unroll 8
  for (int j = 0; j < 32; ++j) {
    int c = l16 + j * 16;
    float v = ldsO[row * DIM + c];
    orow[c] = (v - mu) * rstd * lng[c] + lnb[c] + xr[c];
  }
}

// ---------------------------------------------------------------------------
// Launch
// ---------------------------------------------------------------------------
extern "C" void kernel_launch(void* const* d_in, const int* in_sizes, int n_in,
                              void* d_out, int out_size, void* d_ws, size_t ws_size,
                              hipStream_t stream) {
  const float* x     = (const float*)d_in[0];
  const float* wqkv  = (const float*)d_in[1];
  const float* wout  = (const float*)d_in[2];
  const float* bout  = (const float*)d_in[3];
  const float* lng   = (const float*)d_in[4];
  const float* lnb   = (const float*)d_in[5];
  float* out = (float*)d_out;

  char* ws = (char*)d_ws;                          // ~58 MB total
  __bf16* xbf   = (__bf16*)(ws);                               // 8 MB
  __bf16* wqkvT = (__bf16*)(ws + 8388608);                     // 1.5 MB
  __bf16* woutT = (__bf16*)(ws + 9961472);                     // 0.5 MB
  __bf16* qkv   = (__bf16*)(ws + 10485760);                    // 24 MB
  __bf16* krT   = (__bf16*)(ws + 35651584);                    // 8 MB
  __bf16* vT    = (__bf16*)(ws + 44040192);                    // 8 MB
  __bf16* aoBf  = (__bf16*)(ws + 52428800);                    // 8 MB

  k_prep_x   <<<ROWS * DIM / 256, 256, 0, stream>>>(x, xbf);
  k_prep_wqkv<<<DIM * QKVC / 256, 256, 0, stream>>>(wqkv, wqkvT);
  k_prep_wout<<<DIM * DIM / 256, 256, 0, stream>>>(wout, woutT);

  k_gemm_qkv <<<(ROWS / 16) * 24 / 8, 256, 0, stream>>>(xbf, wqkvT, qkv);
  k_repack_kv<<<64 * NN * DD / 256, 256, 0, stream>>>(qkv, krT, vT);
  k_attn     <<<64 * (NN / 16) / 4, 128, 0, stream>>>(qkv, krT, vT, aoBf);
  k_proj_ln  <<<ROWS / 16, 256, 0, stream>>>(aoBf, woutT, bout, lng, lnb, x, out);
}
